// PCD_Align_34626026340627
// MI455X (gfx1250) — matrix operand order repaired and checked by reading, not compiled, not run on hardware
//
#include <hip/hip_runtime.h>
#include <cstdint>
#include <cstddef>

// ---------------------------------------------------------------------------
// PCD alignment (EDVR) for MI455X / gfx1250.
// Implicit-GEMM convs + DCNv2 on v_wmma_f32_16x16x32_bf16 (wave32).
// K is tap-major (k = tap*Cin + ci): one bounds check / one set of bilinear
// weights per 8-channel block. LDS double-buffered, 1 barrier per 64-k stage.
// ---------------------------------------------------------------------------

typedef __attribute__((ext_vector_type(16))) __bf16          v16bf;
typedef __attribute__((ext_vector_type(16))) unsigned short  v16us;
typedef __attribute__((ext_vector_type(8)))  unsigned short  us8;
typedef __attribute__((ext_vector_type(8)))  float           v8f;

union FragU { struct { us8 lo, hi; } p; v16us v; };

__device__ __forceinline__ unsigned short f2bf(float f) {
  unsigned int u = __float_as_uint(f);
  u += 0x7FFFu + ((u >> 16) & 1u);          // round-to-nearest-even
  return (unsigned short)(u >> 16);
}

__device__ __forceinline__ float lrelu_f(float v) { return v >= 0.f ? v : 0.1f * v; }

// ---- weight repack: OIHW fp32 -> bf16 [Mpad][K], K tap-major --------------
// dst[m][tap*Cin + ci] = w[m][ci][tap]
__global__ void prep_weights_kernel(const float* __restrict__ w,
                                    unsigned short* __restrict__ dst,
                                    int Cout, int Cin, int Mpad) {
  int K = Cin * 9;
  int idx = blockIdx.x * 256 + threadIdx.x;
  if (idx >= Mpad * K) return;
  int m = idx / K;
  int k = idx - m * K;
  int tap = k / Cin;
  int ci  = k - tap * Cin;
  float v = (m < Cout) ? w[(size_t)(m * Cin + ci) * 9 + tap] : 0.f;
  dst[idx] = f2bf(v);
}

// ---- WMMA over one 64-k stage (2 x 16x16x32) ------------------------------
__device__ __forceinline__ v8f wmma_stage(
    const unsigned short (*Atile)[72], const unsigned short (*BtileT)[72],
    int Ml, int Nl, int half, v8f acc) {
#pragma unroll
  for (int s = 0; s < 2; ++s) {
    FragU a, b;
    a.p.lo = *(const us8*)&Atile[Ml][s * 32 + half * 8];
    a.p.hi = *(const us8*)&Atile[Ml][s * 32 + 16 + half * 8];
    b.p.lo = *(const us8*)&BtileT[Nl][s * 32 + half * 8];
    b.p.hi = *(const us8*)&BtileT[Nl][s * 32 + 16 + half * 8];
    acc = __builtin_amdgcn_wmma_f32_16x16x32_bf16(
        false, __builtin_bit_cast(v16bf, a.v),
        false, __builtin_bit_cast(v16bf, b.v),
        (short)0, acc, false, false);
  }
  return acc;
}

// ---- 3x3 conv (pad 1), implicit GEMM, fused concat+bias+lrelu -------------
// Tile 64(M) x 32(N); 8 waves = 4 M-subtiles x 2 N-subtiles.
__global__ __launch_bounds__(256) void conv3x3_wmma_kernel(
    const float* __restrict__ src1, int C1,
    const float* __restrict__ src2, int C2,
    const unsigned short* __restrict__ Abf,   // [Mpad][K] bf16, tap-major
    const float* __restrict__ bias,
    float* __restrict__ out, int Cout,
    int H, int W, int relu) {
  const int Cin    = C1 + C2;                // 64 or 128
  const int K      = Cin * 9;
  const int Ksteps = K >> 6;
  const int lsh    = Cin >> 7;               // 0 (Cin=64) or 1 (Cin=128)

  __shared__ __align__(16) unsigned short Atile[2][64][72];
  __shared__ __align__(16) unsigned short BtileT[2][32][72];

  const int tid  = threadIdx.x;
  const int lane = tid & 31;
  const int wave = tid >> 5;
  const int mrow = wave >> 1;
  const int ncol = wave & 1;
  const int half = lane >> 4;
  const int Ml   = mrow * 16 + (lane & 15);
  const int Nl   = ncol * 16 + (lane & 15);

  const int tileBase = blockIdx.x * 32;      // 32 consecutive pixels, one row
  const int HW   = H * W;
  const int bimg = tileBase / HW;
  const int rem  = tileBase - bimg * HW;
  const int ypix = rem / W;
  const int x0   = rem - ypix * W;

  const int npix = tid >> 3;                 // B loader: pixel 0..31
  const int kq   = (tid & 7) * 8;            // 8 consecutive k (one tap)
  const int xpix = x0 + npix;

  const int arow = tid >> 2;                 // A loader
  const int achk = tid & 3;
  const int gm   = blockIdx.y * 64 + arow;

  auto load_stage = [&](int ks, int buf) {
    // A panel: two b128 global->LDS per thread
    const unsigned short* ap = Abf + (size_t)gm * K + ks * 64;
    *(uint4*)&Atile[buf][arow][achk * 8]      = *(const uint4*)(ap + achk * 8);
    *(uint4*)&Atile[buf][arow][32 + achk * 8] = *(const uint4*)(ap + 32 + achk * 8);
    __builtin_prefetch(ap + 64 + achk * 8, 0, 1);
    // B panel: one tap, 8 channels, unconditional clamped loads
    int tap = ks >> lsh;
    int cb  = ((ks & lsh) << 6) + kq;        // first channel of this block
    int ky  = (tap * 11) >> 5;               // tap/3
    int kx  = tap - ky * 3;
    int yy  = ypix + ky - 1;
    int xx  = xpix + kx - 1;
    float okf = (yy >= 0 && yy < H && xx >= 0 && xx < W) ? 1.f : 0.f;
    int yyc = min(H - 1, max(0, yy));
    int xxc = min(W - 1, max(0, xx));
    const float* sp; int cc, Cs;
    if (cb < C1) { sp = src1; cc = cb;      Cs = C1; }
    else         { sp = src2; cc = cb - C1; Cs = C2; }
    const float* base = sp + ((size_t)(bimg * Cs + cc) * H + yyc) * W + xxc;
    us8 pk;
#pragma unroll
    for (int i = 0; i < 8; ++i)
      pk[i] = f2bf(base[(size_t)i * HW] * okf);
    *(uint4*)&BtileT[buf][npix][kq] = __builtin_bit_cast(uint4, pk);
  };

  v8f acc = {0.f, 0.f, 0.f, 0.f, 0.f, 0.f, 0.f, 0.f};

  load_stage(0, 0);
  __syncthreads();
  int buf = 0;
  for (int ks = 0; ks < Ksteps; ++ks) {
    if (ks + 1 < Ksteps) load_stage(ks + 1, buf ^ 1);
    acc = wmma_stage(Atile[buf], BtileT[buf], Ml, Nl, half, acc);
    __syncthreads();
    buf ^= 1;
  }

  // tile lives in one image row: no division needed for the epilogue decode
  const int x2 = x0 + ncol * 16 + (lane & 15);
#pragma unroll
  for (int r = 0; r < 8; ++r) {
    int m = blockIdx.y * 64 + mrow * 16 + r + half * 8;
    if (m < Cout) {
      float o = acc[r] + bias[m];
      if (relu) o = lrelu_f(o);
      out[((size_t)(bimg * Cout + m) * H + ypix) * W + x2] = o;
    }
  }
}

// ---- modulated deformable conv (DCNv2), implicit GEMM on WMMA -------------
// k = tap*64 + ci; an aligned 8-channel block shares one deformable group,
// so offsets/sigmoid/bilinear weights are computed once per block.
__global__ __launch_bounds__(256) void dcn_wmma_kernel(
    const float* __restrict__ xin,            // [B,64,H,W]
    const float* __restrict__ om,             // [B,216,H,W] raw offset conv out
    const unsigned short* __restrict__ Abf,   // [64][576] bf16, tap-major
    const float* __restrict__ bias,
    float* __restrict__ out,                  // [B,64,H,W] (lrelu fused)
    int H, int W) {
  const int K = 576, Ksteps = 9;              // stage ks <=> tap = ks

  __shared__ __align__(16) unsigned short Atile[2][64][72];
  __shared__ __align__(16) unsigned short BtileT[2][32][72];

  const int tid  = threadIdx.x;
  const int lane = tid & 31;
  const int wave = tid >> 5;
  const int mrow = wave >> 1;
  const int ncol = wave & 1;
  const int half = lane >> 4;
  const int Ml   = mrow * 16 + (lane & 15);
  const int Nl   = ncol * 16 + (lane & 15);

  const int tileBase = blockIdx.x * 32;
  const int HW   = H * W;
  const int bimg = tileBase / HW;
  const int rem  = tileBase - bimg * HW;
  const int ypix = rem / W;
  const int x0   = rem - ypix * W;

  const int npix = tid >> 3;
  const int kq   = (tid & 7) * 8;             // channel block (one group)
  const int xpix = x0 + npix;
  const int grp  = kq >> 3;                   // deformable group 0..7

  const int arow = tid >> 2;
  const int achk = tid & 3;

  const float* omb = om + (size_t)bimg * 216 * HW;
  const float* xpb = xin + (size_t)(bimg * 64 + kq) * HW;  // first channel
  const int rowoff = ypix * W + xpix;

  auto load_stage = [&](int ks, int buf) {
    const unsigned short* ap = Abf + (size_t)arow * K + ks * 64;
    *(uint4*)&Atile[buf][arow][achk * 8]      = *(const uint4*)(ap + achk * 8);
    *(uint4*)&Atile[buf][arow][32 + achk * 8] = *(const uint4*)(ap + 32 + achk * 8);
    __builtin_prefetch(ap + 64 + achk * 8, 0, 1);

    int tap = ks;
    int oc  = grp * 9 + tap;
    int ky  = (tap * 11) >> 5;
    int kx  = tap - ky * 3;
    size_t p = (size_t)oc * HW + rowoff;
    float dy = omb[p];
    float dx = omb[p + (size_t)72  * HW];
    float mm = omb[p + (size_t)144 * HW];
    mm = 1.f / (1.f + __expf(-mm));           // sigmoid mask
    float py  = (float)(ypix + ky - 1) + dy;
    float px  = (float)(xpix + kx - 1) + dx;
    float y0f = floorf(py), x0f = floorf(px);
    float wy = py - y0f, wx = px - x0f;
    int iy = (int)y0f, ix = (int)x0f;
    // validity folded into weights, indices clamped -> unconditional loads
    float vy0 = (iy >= 0 && iy < H) ? 1.f : 0.f;
    float vy1 = (iy + 1 >= 0 && iy + 1 < H) ? 1.f : 0.f;
    float vx0 = (ix >= 0 && ix < W) ? 1.f : 0.f;
    float vx1 = (ix + 1 >= 0 && ix + 1 < W) ? 1.f : 0.f;
    float w00 = (1.f - wy) * (1.f - wx) * vy0 * vx0 * mm;
    float w01 = (1.f - wy) * wx         * vy0 * vx1 * mm;
    float w10 = wy         * (1.f - wx) * vy1 * vx0 * mm;
    float w11 = wy         * wx         * vy1 * vx1 * mm;
    int iy0 = min(H - 1, max(0, iy)), iy1 = min(H - 1, max(0, iy + 1));
    int ix0 = min(W - 1, max(0, ix)), ix1 = min(W - 1, max(0, ix + 1));
    int o00 = iy0 * W + ix0, o01 = iy0 * W + ix1;
    int o10 = iy1 * W + ix0, o11 = iy1 * W + ix1;
    us8 pk;
#pragma unroll
    for (int i = 0; i < 8; ++i) {
      const float* cp = xpb + (size_t)i * HW;
      float v = cp[o00] * w00 + cp[o01] * w01 + cp[o10] * w10 + cp[o11] * w11;
      pk[i] = f2bf(v);
    }
    *(uint4*)&BtileT[buf][npix][kq] = __builtin_bit_cast(uint4, pk);
  };

  v8f acc = {0.f, 0.f, 0.f, 0.f, 0.f, 0.f, 0.f, 0.f};

  load_stage(0, 0);
  __syncthreads();
  int buf = 0;
  for (int ks = 0; ks < Ksteps; ++ks) {
    if (ks + 1 < Ksteps) load_stage(ks + 1, buf ^ 1);
    acc = wmma_stage(Atile[buf], BtileT[buf], Ml, Nl, half, acc);
    __syncthreads();
    buf ^= 1;
  }

  const int x2 = x0 + ncol * 16 + (lane & 15);
#pragma unroll
  for (int r = 0; r < 8; ++r) {
    int m = mrow * 16 + r + half * 8;
    float o = lrelu_f(acc[r] + bias[m]);
    out[((size_t)(bimg * 64 + m) * H + ypix) * W + x2] = o;
  }
}

// ---- bilinear x2 upsample, half-pixel centers, fused scale ----------------
__global__ void up2_kernel(const float* __restrict__ in, float* __restrict__ out,
                           int C, int H, int W, float scale) {
  int Ho = 2 * H, Wo = 2 * W;
  long total = 4L * C * Ho * Wo;
  long idx = (long)blockIdx.x * 256 + threadIdx.x;
  if (idx >= total) return;
  int xo = (int)(idx % Wo); long t = idx / Wo;
  int yo = (int)(t % Ho);   t /= Ho;
  int c  = (int)(t % C);
  int b  = (int)(t / C);
  float sy = (yo + 0.5f) * 0.5f - 0.5f;
  float sx = (xo + 0.5f) * 0.5f - 0.5f;
  float y0f = floorf(sy), x0f = floorf(sx);
  float wy = sy - y0f, wx = sx - x0f;
  int iy0 = min(H - 1, max(0, (int)y0f));
  int iy1 = min(H - 1, max(0, (int)y0f + 1));
  int ix0 = min(W - 1, max(0, (int)x0f));
  int ix1 = min(W - 1, max(0, (int)x0f + 1));
  const float* p = in + (size_t)(b * C + c) * H * W;
  float v = p[(size_t)iy0 * W + ix0] * (1.f - wy) * (1.f - wx)
          + p[(size_t)iy0 * W + ix1] * (1.f - wy) * wx
          + p[(size_t)iy1 * W + ix0] * wy * (1.f - wx)
          + p[(size_t)iy1 * W + ix1] * wy * wx;
  out[idx] = v * scale;
}

// ---------------------------------------------------------------------------
extern "C" void kernel_launch(void* const* d_in, const int* in_sizes, int n_in,
                              void* d_out, int out_size, void* d_ws, size_t ws_size,
                              hipStream_t stream) {
  const int B = 4, H1 = 128, W1 = 128, H2 = 64, W2 = 64, H3 = 32, W3 = 32;

  enum { L3_OC1 = 0, L3_OC2, L3_OM, L3_DCN,
         L2_OC1, L2_OC2, L2_OC3, L2_OM, L2_DCN, L2_FC,
         L1_OC1, L1_OC2, L1_OC3, L1_OM, L1_DCN, L1_FC,
         CAS_OC1, CAS_OC2, CAS_OM, CAS_DCN, NW };
  struct WInfo { int widx, bidx, Cin, Cout, Mpad; };
  const WInfo WT[NW] = {
    {6, 7, 128, 64, 64},  {8, 9, 64, 64, 64},   {10, 11, 64, 216, 256}, {12, 13, 64, 64, 64},
    {14, 15, 128, 64, 64},{16, 17, 128, 64, 64},{18, 19, 64, 64, 64},   {20, 21, 64, 216, 256},
    {22, 23, 64, 64, 64}, {24, 25, 128, 64, 64},
    {26, 27, 128, 64, 64},{28, 29, 128, 64, 64},{30, 31, 64, 64, 64},   {32, 33, 64, 216, 256},
    {34, 35, 64, 64, 64}, {36, 37, 128, 64, 64},
    {38, 39, 128, 64, 64},{40, 41, 64, 64, 64}, {42, 43, 64, 216, 256}, {44, 45, 64, 64, 64},
  };

  size_t woff[NW], cur = 0;
  for (int i = 0; i < NW; ++i) { woff[i] = cur; cur += (size_t)WT[i].Mpad * WT[i].Cin * 9; }
  unsigned short* wbase = (unsigned short*)d_ws;
  size_t wbytes = (cur * sizeof(unsigned short) + 255) & ~(size_t)255;
  float* fbase = (float*)((char*)d_ws + wbytes);

  const size_t P1 = (size_t)B * 64 * H1 * W1;
  const size_t P2 = (size_t)B * 64 * H2 * W2;
  const size_t P3 = (size_t)B * 64 * H3 * W3;
  const size_t POM = (size_t)B * 216 * H1 * W1;
  size_t o = 0;
  float* BUF_A = fbase + o; o += P1;
  float* BUF_B = fbase + o; o += P1;
  float* UP    = fbase + o; o += P1;
  float* L3OFF = fbase + o; o += P3;
  float* L3FEA = fbase + o; o += P3;
  float* L2OFF = fbase + o; o += P2;
  float* L2FEA = fbase + o; o += P2;
  float* L1FEA = fbase + o; o += P1;
  float* OM    = fbase + o; o += POM;
  (void)ws_size; (void)in_sizes; (void)n_in; (void)out_size;

  for (int i = 0; i < NW; ++i) {
    int tot = WT[i].Mpad * WT[i].Cin * 9;
    prep_weights_kernel<<<(tot + 255) / 256, 256, 0, stream>>>(
        (const float*)d_in[WT[i].widx], wbase + woff[i],
        WT[i].Cout, WT[i].Cin, WT[i].Mpad);
  }

  const float* nbr1 = (const float*)d_in[0];
  const float* nbr2 = (const float*)d_in[1];
  const float* nbr3 = (const float*)d_in[2];
  const float* ref1 = (const float*)d_in[3];
  const float* ref2 = (const float*)d_in[4];
  const float* ref3 = (const float*)d_in[5];

  auto wp = [&](int i) { return (const unsigned short*)(wbase + woff[i]); };
  auto bp = [&](int i) { return (const float*)d_in[WT[i].bidx]; };
  auto conv = [&](const float* s1, int C1, const float* s2, int C2, int wi,
                  float* dst, int H, int W, int relu) {
    dim3 g((unsigned)((B * H * W) / 32), (unsigned)(WT[wi].Mpad / 64));
    conv3x3_wmma_kernel<<<g, 256, 0, stream>>>(s1, C1, s2, C2, wp(wi), bp(wi),
                                               dst, WT[wi].Cout, H, W, relu);
  };
  auto dcn = [&](const float* x, const float* omp, int wi, float* dst, int H, int W) {
    dim3 g((unsigned)((B * H * W) / 32), 1);
    dcn_wmma_kernel<<<g, 256, 0, stream>>>(x, omp, wp(wi), bp(wi), dst, H, W);
  };
  auto up2 = [&](const float* in, float* dst, int H, int W, float s) {
    long tot = (long)B * 64 * 4 * H * W;
    up2_kernel<<<(unsigned)((tot + 255) / 256), 256, 0, stream>>>(in, dst, 64, H, W, s);
  };

  // ---- L3 ----
  conv(nbr3, 64, ref3, 64, L3_OC1, BUF_A, H3, W3, 1);
  conv(BUF_A, 64, nullptr, 0, L3_OC2, L3OFF, H3, W3, 1);
  conv(L3OFF, 64, nullptr, 0, L3_OM, OM, H3, W3, 0);
  dcn(nbr3, OM, L3_DCN, L3FEA, H3, W3);
  // ---- L2 ----
  conv(nbr2, 64, ref2, 64, L2_OC1, BUF_A, H2, W2, 1);
  up2(L3OFF, UP, H3, W3, 2.f);
  conv(BUF_A, 64, UP, 64, L2_OC2, BUF_B, H2, W2, 1);
  conv(BUF_B, 64, nullptr, 0, L2_OC3, L2OFF, H2, W2, 1);
  conv(L2OFF, 64, nullptr, 0, L2_OM, OM, H2, W2, 0);
  dcn(nbr2, OM, L2_DCN, BUF_A, H2, W2);
  up2(L3FEA, UP, H3, W3, 1.f);
  conv(BUF_A, 64, UP, 64, L2_FC, L2FEA, H2, W2, 1);
  // ---- L1 ----
  conv(nbr1, 64, ref1, 64, L1_OC1, BUF_A, H1, W1, 1);
  up2(L2OFF, UP, H2, W2, 2.f);
  conv(BUF_A, 64, UP, 64, L1_OC2, BUF_B, H1, W1, 1);
  conv(BUF_B, 64, nullptr, 0, L1_OC3, BUF_A, H1, W1, 1);
  conv(BUF_A, 64, nullptr, 0, L1_OM, OM, H1, W1, 0);
  dcn(nbr1, OM, L1_DCN, BUF_B, H1, W1);
  up2(L2FEA, UP, H2, W2, 1.f);
  conv(BUF_B, 64, UP, 64, L1_FC, L1FEA, H1, W1, 0);   // no lrelu (matches ref)
  // ---- cascading refinement ----
  conv(L1FEA, 64, ref1, 64, CAS_OC1, BUF_A, H1, W1, 1);
  conv(BUF_A, 64, nullptr, 0, CAS_OC2, BUF_B, H1, W1, 1);
  conv(BUF_B, 64, nullptr, 0, CAS_OM, OM, H1, W1, 0);
  dcn(L1FEA, OM, CAS_DCN, (float*)d_out, H1, W1);
}
